// TwoHeadedLSTMQNetwork_49959059587607
// MI455X (gfx1250) — matrix operand order repaired and checked by reading, not compiled, not run on hardware
//
#include <hip/hip_runtime.h>
#include <hip/hip_bf16.h>
#include <stdint.h>

// ---------------------------------------------------------------------------
// Two-headed LSTM Q-network for MI455X (gfx1250, wave32, WMMA f16 16x16x32).
//
//   setup_kernel : f32->f16 convert obses; pack [Wx;Wh] (640x2048) and Wv
//                  (512x48, zero-padded) into per-lane WMMA B-fragment layout;
//                  init xh = [x_0 | h0], c-state = c0, zero grid barrier.
//   lstm_kernel  : persistent, 64 blocks x 256 thr = 512 waves; one 16x16
//                  Z-tile per wave per step. Weights (20 B-fragments) held in
//                  VGPRs across all 1024 steps; A row-tile staged in LDS per
//                  block per step; 20x v_wmma_f32_16x16x32_f16 over K=640;
//                  grid barrier; fused sigmoid/tanh gate update; grid barrier.
//   qproj_kernel : (65536x512 f16) @ (512x48 f16) WMMA, masked store q0/q1.
//
// Workspace: ~87.3 MB required.
// ---------------------------------------------------------------------------

typedef __attribute__((ext_vector_type(16))) _Float16 v16h;
typedef __attribute__((ext_vector_type(8)))  _Float16 v8h;
typedef __attribute__((ext_vector_type(8)))  float    v8f;

#define B_    64
#define T_    1024
#define D_    128
#define H_    512
#define G_    2048          // 4*H
#define KXH   640           // D + H
#define KS_XH 20            // 640/32 k-steps
#define NT_XH 128           // 2048/16 col tiles
#define A_    18
#define NQ    48            // padded 2*A
#define NT_Q  3
#define KS_Q  16            // 512/32

// ---- WMMA fragment loaders (layouts per CDNA5 ISA 7.12.2) -------

// A (16x32 f16): lane L holds row M=L&15; half=L>>4 selects K banks:
//   elem e in [0,8)  -> K = 8*half + e
//   elem e in [8,16) -> K = 8*half + 16 + (e-8)
// Works for both global and LDS (generic) pointers.
__device__ inline v16h load_a_frag(const _Float16* row, int half) {
  const v8h lo = *(const v8h*)(row + 8 * half);
  const v8h hi = *(const v8h*)(row + 8 * half + 16);
  v16h r;
#pragma unroll
  for (int i = 0; i < 8; ++i) { r[i] = lo[i]; r[i + 8] = hi[i]; }
  return r;
}

// B (32x16 f16): pre-packed so each lane's 16 elements are contiguous (32B).
__device__ inline v16h load_b_frag(const _Float16* __restrict__ p) {
  const v8h lo = *(const v8h*)(p);
  const v8h hi = *(const v8h*)(p + 8);
  v16h r;
#pragma unroll
  for (int i = 0; i < 8; ++i) { r[i] = lo[i]; r[i + 8] = hi[i]; }
  return r;
}

__device__ inline float sigmoidf_(float x) { return 1.0f / (1.0f + __expf(-x)); }

// Grid-wide barrier: monotonic L2 atomic counter, s_sleep spin. Requires all
// blocks resident (64 blocks << WGP count on MI455X). Counter re-zeroed by
// setup_kernel each launch -> deterministic across graph replays.
__device__ inline void grid_barrier(unsigned* bar, unsigned target) {
  __threadfence();
  __syncthreads();
  if (threadIdx.x == 0) {
    __hip_atomic_fetch_add(bar, 1u, __ATOMIC_ACQ_REL, __HIP_MEMORY_SCOPE_AGENT);
    while (__hip_atomic_load(bar, __ATOMIC_ACQUIRE, __HIP_MEMORY_SCOPE_AGENT) < target) {
      __builtin_amdgcn_s_sleep(2);
    }
  }
  __syncthreads();
}

// ---------------------------------------------------------------------------
__global__ __launch_bounds__(256) void setup_kernel(
    const float* __restrict__ obses, const float* __restrict__ c0,
    const float* __restrict__ h0,    const float* __restrict__ Wx,
    const float* __restrict__ Wh,    const float* __restrict__ Wv,
    _Float16* __restrict__ obsF16,   _Float16* __restrict__ packWxh,
    _Float16* __restrict__ packWv,   _Float16* __restrict__ xh,
    float* __restrict__ cst,         unsigned* __restrict__ bar) {
  const size_t N1 = (size_t)B_ * T_ * D_;            // obses f16 convert
  const size_t N2 = (size_t)NT_XH * KS_XH * 32 * 16; // packed [Wx;Wh]
  const size_t N3 = (size_t)NT_Q * KS_Q * 32 * 16;   // packed Wv
  const size_t N4 = (size_t)B_ * KXH;                // xh init
  const size_t N5 = (size_t)B_ * H_;                 // c state init
  const size_t total = N1 + N2 + N3 + N4 + N5 + 1;
  const size_t stride = (size_t)gridDim.x * blockDim.x;
  for (size_t i = (size_t)blockIdx.x * blockDim.x + threadIdx.x; i < total; i += stride) {
    if (i < N1) {
      obsF16[i] = (_Float16)obses[i];
    } else if (i < N1 + N2) {
      size_t idx = i - N1;                      // ((ct*20+ks)*32+lane)*16+e
      int e = (int)(idx & 15), lane = (int)((idx >> 4) & 31);
      int rest = (int)(idx >> 9);
      int ks = rest % KS_XH, ct = rest / KS_XH;
      int k = ks * 32 + 8 * (lane >> 4) + (e & 7) + 16 * (e >> 3);   // B frag: K = 8*half + (e&7) + 16*(e>>3)
      int g = ct * 16 + (lane & 15);            //         N = lane&15
      float w = (k < D_) ? Wx[(size_t)k * G_ + g] : Wh[(size_t)(k - D_) * G_ + g];
      packWxh[idx] = (_Float16)w;
    } else if (i < N1 + N2 + N3) {
      size_t idx = i - (N1 + N2);
      int e = (int)(idx & 15), lane = (int)((idx >> 4) & 31);
      int rest = (int)(idx >> 9);
      int ks = rest % KS_Q, ct = rest / KS_Q;
      int k = ks * 32 + 8 * (lane >> 4) + (e & 7) + 16 * (e >> 3);
      int c = ct * 16 + (lane & 15);
      packWv[idx] = (c < 2 * A_) ? (_Float16)Wv[(size_t)k * (2 * A_) + c] : (_Float16)0.0f;
    } else if (i < N1 + N2 + N3 + N4) {
      size_t idx = i - (N1 + N2 + N3);
      int bb = (int)(idx / KXH), j = (int)(idx % KXH);
      xh[idx] = (j < D_) ? (_Float16)obses[((size_t)bb * T_ + 0) * D_ + j]
                         : (_Float16)h0[(size_t)bb * H_ + (j - D_)];
    } else if (i < N1 + N2 + N3 + N4 + N5) {
      size_t idx = i - (N1 + N2 + N3 + N4);
      cst[idx] = c0[idx];
    } else {
      *bar = 0u;
    }
  }
}

// ---------------------------------------------------------------------------
// Persistent recurrence kernel. Grid MUST be 64 blocks x 256 threads.
// All 8 waves of a block share one A row-tile (rt = blockIdx>>4) -> stage it
// in LDS once per step. B fragments are time-invariant -> held in VGPRs.
__global__ __launch_bounds__(256) void lstm_kernel(
    const _Float16* __restrict__ obsF16, const _Float16* __restrict__ packWxh,
    const float* __restrict__ bias,      _Float16* __restrict__ xh,
    float* __restrict__ Z,               float* __restrict__ cst,
    _Float16* __restrict__ hist,         float* __restrict__ outC,
    float* __restrict__ outH,            unsigned* __restrict__ bar) {
  __shared__ __align__(16) _Float16 sA[16 * KXH];  // 20 KB: 16 rows x 640

  const int tid  = threadIdx.x;
  const int lane = tid & 31;
  const int wid  = blockIdx.x * 8 + (tid >> 5);  // 0..511
  const int rt   = blockIdx.x >> 4;              // row tile 0..3 (== wid>>7)
  const int ct   = wid & 127;                    // col tile 0..127
  const int nloc = lane & 15;
  const int half = lane >> 4;
  const int gcol = ct * 16 + nloc;
  const float bcol = bias[gcol];
  const int gtid = blockIdx.x * 256 + tid;       // 0..16383
  const unsigned nb = gridDim.x;
  unsigned barNum = 0;

  // Hoist the 20 time-invariant B fragments into VGPRs (160 VGPRs/lane).
  const _Float16* bbase = packWxh + ((size_t)(ct * KS_XH) * 32 + lane) * 16;
  v16h Breg[KS_XH];
#pragma unroll
  for (int ks = 0; ks < KS_XH; ++ks) Breg[ks] = load_b_frag(bbase + (size_t)ks * 512);

  const _Float16* aLds = sA + (size_t)nloc * KXH;

  for (int t = 0; t < T_; ++t) {
    // ---- stage A row-tile [x_t | h] (16 x 640 f16) into LDS ----
#pragma unroll
    for (int cchunk = 0; cchunk < 5; ++cchunk) {
      int chunk = tid + cchunk * 256;            // 0..1279 (8-elem chunks)
      int row = chunk / 80;
      int c8  = (chunk % 80) * 8;
      *(v8h*)(sA + row * KXH + c8) =
          *(const v8h*)(xh + (size_t)(rt * 16 + row) * KXH + c8);
    }
    __syncthreads();

    // ---- z = [x_t | h] @ [Wx;Wh] + b : 20 WMMAs over K=640 ----
    v8f acc;
#pragma unroll
    for (int r = 0; r < 8; ++r) acc[r] = bcol;
#pragma unroll
    for (int ks = 0; ks < KS_XH; ++ks) {
      v16h a = load_a_frag(aLds + ks * 32, half);
      acc = __builtin_amdgcn_wmma_f32_16x16x32_f16(false, a, false, Breg[ks],
                                                   (short)0, acc, false, false);
    }
    {  // scatter tile to Z (D layout: M = r + 8*half, N = lane&15)
      float* zp = Z + (size_t)(rt * 16 + 8 * half) * G_ + gcol;
#pragma unroll
      for (int r = 0; r < 8; ++r) zp[(size_t)r * G_] = acc[r];
    }
    ++barNum;
    grid_barrier(bar, barNum * nb);

    // ---- fused gate update: 32768 (b,j) elems over 16384 threads ----
#pragma unroll
    for (int k = 0; k < 2; ++k) {
      int idx = gtid + k * 16384;
      int bb = idx >> 9, j = idx & 511;
      const float* zr = Z + (size_t)bb * G_;
      float zi = zr[j], zf = zr[H_ + j], zg = zr[2 * H_ + j], zo = zr[3 * H_ + j];
      float c = cst[idx];
      c = sigmoidf_(zf) * c + sigmoidf_(zi) * tanhf(zg);
      float h = sigmoidf_(zo) * tanhf(c);
      cst[idx] = c;
      _Float16 hf = (_Float16)h;
      xh[(size_t)bb * KXH + D_ + j] = hf;                 // next-step A rows
      hist[((size_t)t * B_ + bb) * H_ + j] = hf;          // for head GEMM
      if (t == T_ - 1) { outC[idx] = c; outH[idx] = h; }  // final carry
    }
    // refresh x part of xh for step t+1 (+ prefetch t+2 slice toward WGP)
    if (t < T_ - 1 && gtid < B_ * D_) {
      int bb = gtid >> 7, d = gtid & 127;
      const _Float16* src = obsF16 + ((size_t)bb * T_ + (t + 1)) * D_ + d;
      xh[(size_t)bb * KXH + d] = *src;
      __builtin_prefetch(src + D_, 0, 3);
    }
    ++barNum;
    grid_barrier(bar, barNum * nb);
  }
}

// ---------------------------------------------------------------------------
// Head projection: (65536 x 512 f16 hist) @ (512 x 48 packed Wv) -> q0, q1.
// Grid MUST be 512 blocks x 256 threads (4096 waves = 4096 row tiles).
__global__ __launch_bounds__(256) void qproj_kernel(
    const _Float16* __restrict__ hist, const _Float16* __restrict__ packWv,
    const float* __restrict__ bv, float* __restrict__ q0, float* __restrict__ q1) {
  const int lane = threadIdx.x & 31;
  const int rt   = blockIdx.x * 8 + (threadIdx.x >> 5);  // row tile 0..4095
  const int nloc = lane & 15;
  const int half = lane >> 4;
  const _Float16* arow = hist + (size_t)(rt * 16 + nloc) * H_;
  for (int ctl = 0; ctl < NT_Q; ++ctl) {
    const int c = ctl * 16 + nloc;
    const float bc = (c < 2 * A_) ? bv[c] : 0.0f;
    v8f acc;
#pragma unroll
    for (int r = 0; r < 8; ++r) acc[r] = bc;
    const _Float16* bp = packWv + ((size_t)(ctl * KS_Q) * 32 + lane) * 16;
#pragma unroll 4
    for (int ks = 0; ks < KS_Q; ++ks) {
      v16h a = load_a_frag(arow + ks * 32, half);
      v16h b = load_b_frag(bp + (size_t)ks * 512);
      acc = __builtin_amdgcn_wmma_f32_16x16x32_f16(false, a, false, b,
                                                   (short)0, acc, false, false);
    }
#pragma unroll
    for (int r = 0; r < 8; ++r) {
      int R = rt * 16 + r + 8 * half;  // global row = t*64 + b
      int tt = R >> 6, bb = R & 63;
      float v = acc[r];
      if (c < A_)            q0[((size_t)bb * T_ + tt) * A_ + c] = v;
      else if (c < 2 * A_)   q1[((size_t)bb * T_ + tt) * A_ + (c - A_)] = v;
    }
  }
}

// ---------------------------------------------------------------------------
extern "C" void kernel_launch(void* const* d_in, const int* in_sizes, int n_in,
                              void* d_out, int out_size, void* d_ws, size_t ws_size,
                              hipStream_t stream) {
  (void)in_sizes; (void)n_in; (void)out_size;
  const float* obses = (const float*)d_in[0];
  const float* c0    = (const float*)d_in[1];
  const float* h0    = (const float*)d_in[2];
  const float* Wx    = (const float*)d_in[3];
  const float* Wh    = (const float*)d_in[4];
  const float* b     = (const float*)d_in[5];
  const float* Wv    = (const float*)d_in[6];
  const float* bv    = (const float*)d_in[7];
  float* out = (float*)d_out;

  // workspace carve-up (256B aligned), total ~87.3 MB
  uint8_t* ws = (uint8_t*)d_ws;
  size_t off = 0;
  auto carve = [&](size_t bytes) -> uint8_t* {
    uint8_t* p = ws + off;
    off = (off + bytes + 255u) & ~(size_t)255u;
    return p;
  };
  _Float16* obsF16  = (_Float16*)carve((size_t)B_ * T_ * D_ * 2);      // 16 MB
  _Float16* packWxh = (_Float16*)carve((size_t)KXH * G_ * 2);          // 2.5 MB
  _Float16* packWv  = (_Float16*)carve((size_t)H_ * NQ * 2);           // 48 KB
  _Float16* xh      = (_Float16*)carve((size_t)B_ * KXH * 2);          // 80 KB
  float*    Z       = (float*)carve((size_t)B_ * G_ * 4);              // 512 KB
  float*    cst     = (float*)carve((size_t)B_ * H_ * 4);              // 128 KB
  _Float16* hist    = (_Float16*)carve((size_t)T_ * B_ * H_ * 2);      // 64 MB
  unsigned* bar     = (unsigned*)carve(256);
  (void)ws_size;  // requires >= ~87.3 MB

  float* outC = out;                         // carry.c  (64x512)
  float* outH = out + (size_t)B_ * H_;       // carry.h  (64x512)
  float* q0   = out + (size_t)2 * B_ * H_;   // (64,1024,18)
  float* q1   = q0 + (size_t)B_ * T_ * A_;   // (64,1024,18)

  setup_kernel<<<8192, 256, 0, stream>>>(obses, c0, h0, Wx, Wh, Wv,
                                         obsF16, packWxh, packWv, xh, cst, bar);
  lstm_kernel<<<64, 256, 0, stream>>>(obsF16, packWxh, b, xh, Z, cst, hist,
                                      outC, outH, bar);
  qproj_kernel<<<512, 256, 0, stream>>>(hist, packWv, bv, q0, q1);
}
